// tinyBlock_36455682408714
// MI455X (gfx1250) — matrix operand order repaired and checked by reading, not compiled, not run on hardware
//
#include <hip/hip_runtime.h>
#include <hip/hip_bf16.h>

// ---------------------------------------------------------------------------
// Tiny transformer block (B=4, T=4096, D=256) for gfx1250 (MI455X).
// All GEMMs use v_wmma_f32_16x16x32_f16. Activations kept in f16 in the
// workspace (L2-resident: 192MB L2 >> ~100MB working set).
// ---------------------------------------------------------------------------

typedef __attribute__((ext_vector_type(16))) _Float16 v16h;
typedef __attribute__((ext_vector_type(8)))  float    v8f;

union Frag {
    v16h h;
    uint4 u[2];
};

#define WMMA_F16(a, b, c) \
    __builtin_amdgcn_wmma_f32_16x16x32_f16(false, (a), false, (b), (short)0, (c), false, false)

static constexpr int BB = 4;       // batch
static constexpr int TT = 4096;    // seq len
static constexpr int DD = 256;     // model dim
static constexpr int HH = 1024;    // mlp hidden
static constexpr int NT = BB * TT; // 16384 token rows

// ---------------------------------------------------------------------------
// fp32 -> fp16 weight conversion (grid-stride)
// ---------------------------------------------------------------------------
__global__ __launch_bounds__(256) void cvt_f32_to_f16(const float* __restrict__ src,
                                                      _Float16* __restrict__ dst, int n) {
    for (int i = blockIdx.x * blockDim.x + threadIdx.x; i < n; i += gridDim.x * blockDim.x)
        dst[i] = (_Float16)src[i];
}

// ---------------------------------------------------------------------------
// RMSNorm: one 256-thread block per token row; writes f16 normalized row.
// ---------------------------------------------------------------------------
__global__ __launch_bounds__(256) void rmsnorm_f16(const float* __restrict__ x,
                                                   const float* __restrict__ w,
                                                   _Float16* __restrict__ out) {
    __shared__ float red[256];
    const int row = blockIdx.x;
    const int tid = threadIdx.x;
    const float v = x[(size_t)row * DD + tid];
    red[tid] = v * v;
    for (int s = 128; s > 0; s >>= 1) {
        __syncthreads();
        if (tid < s) red[tid] += red[tid + s];
    }
    __syncthreads();
    const float inv = rsqrtf(red[0] * (1.0f / DD) + 1e-6f);
    out[(size_t)row * DD + tid] = (_Float16)(v * inv * w[tid]);
}

// ---------------------------------------------------------------------------
// Generic GEMM: out = epilogue(A[M,K](f16) * W[N,K](f16)^T + bias).
// One wave computes a 32x64 output tile (2 A-fragments x 4 B-fragments,
// 8 WMMA accumulators = 64 VGPRs; 12 b128 loads per 8 WMMAs).
// K is a template parameter so the k-loop unrolls and loads pipeline.
// MODE 0: out_h[m*N+n]           = val                    (q, k)
// MODE 1: out_h[(b*N+n)*T + t]   = val   (V, transposed)  (v)
// MODE 2: out_f[m*N+n]           = resid + val            (o-proj / final)
// MODE 3: out_h[m*N+n]           = gelu(val)              (mlp up)
// ---------------------------------------------------------------------------
template <int MODE, int K>
__global__ __launch_bounds__(32) void gemm_wmma(const _Float16* __restrict__ A,
                                                const _Float16* __restrict__ W,
                                                const float* __restrict__ bias,
                                                const float* __restrict__ resid,
                                                _Float16* __restrict__ out_h,
                                                float* __restrict__ out_f,
                                                int N) {
    const int lane = threadIdx.x;
    const int mrow = lane & 15;
    const int half = lane >> 4;
    const int row0 = blockIdx.x * 32;
    const int n0   = blockIdx.y * 64;

    v8f acc[2][4] = {};
    const _Float16* arow0 = A + (size_t)(row0 + mrow) * K;
    const _Float16* arow1 = arow0 + (size_t)16 * K;

#pragma unroll 8
    for (int k0 = 0; k0 < K; k0 += 32) {
        Frag a0, a1;
        a0.u[0] = *(const uint4*)(arow0 + k0 + half * 8);
        a0.u[1] = *(const uint4*)(arow0 + k0 + 16 + half * 8);
        a1.u[0] = *(const uint4*)(arow1 + k0 + half * 8);
        a1.u[1] = *(const uint4*)(arow1 + k0 + 16 + half * 8);

        const _Float16* wbase = W + (size_t)(n0 + mrow) * K + k0 + half * 8;
#pragma unroll
        for (int j = 0; j < 4; ++j) {
            Frag b;
            b.u[0] = *(const uint4*)(wbase + (size_t)(j * 16) * K);
            b.u[1] = *(const uint4*)(wbase + (size_t)(j * 16) * K + 16);
            acc[0][j] = WMMA_F16(a0.h, b.h, acc[0][j]);
            acc[1][j] = WMMA_F16(a1.h, b.h, acc[1][j]);
        }
    }

#pragma unroll
    for (int mt = 0; mt < 2; ++mt) {
#pragma unroll
        for (int j = 0; j < 4; ++j) {
            const int n = n0 + j * 16 + mrow;
            const float bval = bias[n];
#pragma unroll
            for (int v = 0; v < 8; ++v) {
                const int m = row0 + mt * 16 + v + 8 * half;   // global token row
                const float val = acc[mt][j][v] + bval;
                if (MODE == 0) {
                    out_h[(size_t)m * N + n] = (_Float16)val;
                } else if (MODE == 1) {
                    const int b = m >> 12;                     // m / T
                    const int t = m & (TT - 1);
                    out_h[((size_t)b * N + n) * TT + t] = (_Float16)val;
                } else if (MODE == 2) {
                    const size_t idx = (size_t)m * N + n;
                    out_f[idx] = resid[idx] + val;
                } else { // MODE 3: exact GELU
                    const float g = 0.5f * val * (1.0f + erff(val * 0.70710678118654752f));
                    out_h[(size_t)m * N + n] = (_Float16)g;
                }
            }
        }
    }
}

// ---------------------------------------------------------------------------
// Cooperative 2-wave flash-attention step for one 16-row query block,
// 32 keys per step.  Wave w:
//   - computes score tile w (16x16) = Q(16x256) * K[s0+16w..]^T  (8 WMMAs)
//   - exchanges row maxima via LDS (barrier 1)
//   - exp(), stages its 16 P columns (f16) + partial row sums    (barrier 2)
//   - PV on its 128-feature half: acc(16x128) += P(16x32)*V      (8 WMMAs)
// No duplicated math; per-wave regs ~ acc(64)+Q(64) => slack for the
// scheduler to keep several B-fragment loads in flight.
// ---------------------------------------------------------------------------
template <bool MASK>
__device__ __forceinline__ void attn_step(int s0, int t0, int w, int mrow, int half,
                                          const _Float16* __restrict__ qrow,
                                          const _Float16* __restrict__ kbase,
                                          const _Float16* __restrict__ vbase,
                                          _Float16* pstage, float* redmax, float* redsum,
                                          v8f (&acc)[8], float (&rmax)[8], float (&lsum)[8]) {
    // ---- my score tile: keys [s0 + 16w, s0 + 16w + 16) ----
    v8f sc = {};
    {
        const _Float16* krow = kbase + (size_t)(s0 + w * 16 + mrow) * DD;
        Frag qa, bf;
        qa.u[0] = *(const uint4*)(qrow + half * 8);
        qa.u[1] = *(const uint4*)(qrow + 16 + half * 8);
        bf.u[0] = *(const uint4*)(krow + half * 8);
        bf.u[1] = *(const uint4*)(krow + 16 + half * 8);
#pragma unroll
        for (int kk = 0; kk < 8; ++kk) {
            Frag qn, bn;
            if (kk < 7) {   // issue next loads before consuming current frags
                qn.u[0] = *(const uint4*)(qrow + (kk + 1) * 32 + half * 8);
                qn.u[1] = *(const uint4*)(qrow + (kk + 1) * 32 + 16 + half * 8);
                bn.u[0] = *(const uint4*)(krow + (kk + 1) * 32 + half * 8);
                bn.u[1] = *(const uint4*)(krow + (kk + 1) * 32 + 16 + half * 8);
            }
            sc = WMMA_F16(qa.h, bf.h, sc);
            if (kk < 7) { qa = qn; bf = bn; }
        }
    }

    // ---- scale + (optional) mask + within-tile row max; publish maxima ----
    float sv[8], mx[8];
#pragma unroll
    for (int v = 0; v < 8; ++v) {
        float s = sc[v] * 0.0625f;
        if (MASK) {
            const int mg = t0 + v + 8 * half;               // query row (in-batch)
            s = ((s0 + w * 16 + mrow) > mg) ? -1.0e30f : s;
        }
        float m = s;
#pragma unroll
        for (int off = 1; off < 16; off <<= 1)
            m = fmaxf(m, __shfl_xor(m, off));                // within 16-lane half
        sv[v] = s;
        mx[v] = m;
        if (mrow == v) redmax[w * 16 + v + 8 * half] = m;    // one lane per row
    }
    __syncthreads();   // barrier 1: maxima visible

    // ---- combine maxima, exp, stage P + partial sums, rescale acc ----
#pragma unroll
    for (int v = 0; v < 8; ++v) {
        const float omax = redmax[(1 - w) * 16 + v + 8 * half];
        const float mnew = fmaxf(rmax[v], fmaxf(mx[v], omax));
        const float rescale = __expf(rmax[v] - mnew);
        rmax[v] = mnew;

        const float p = __expf(sv[v] - mnew);
        float ps = p;
#pragma unroll
        for (int off = 1; off < 16; off <<= 1)
            ps += __shfl_xor(ps, off);
        lsum[v] = lsum[v] * rescale + ps;
        if (mrow == v) redsum[w * 16 + v + 8 * half] = ps;

#pragma unroll
        for (int j = 0; j < 8; ++j) acc[j][v] *= rescale;

        // stage my 16 P columns (C-layout -> row-major 16x32 f16)
        pstage[(v + 8 * half) * 32 + w * 16 + mrow] = (_Float16)p;
    }
    __syncthreads();   // barrier 2: P + sums visible

    // ---- add the other wave's partial row sums ----
#pragma unroll
    for (int v = 0; v < 8; ++v)
        lsum[v] += redsum[(1 - w) * 16 + v + 8 * half];

    // ---- reload full P as A-operand fragment ----
    Frag pf;
    const _Float16* prow = pstage + mrow * 32;
    pf.u[0] = *(const uint4*)(prow + half * 8);
    pf.u[1] = *(const uint4*)(prow + 16 + half * 8);

    // ---- acc += P * V on my 128-feature half (V transposed) ----
    Frag bf;
    {
        const _Float16* vr = vbase + (size_t)mrow * TT + s0 + half * 8;
        bf.u[0] = *(const uint4*)(vr);
        bf.u[1] = *(const uint4*)(vr + 16);
    }
#pragma unroll
    for (int j = 0; j < 8; ++j) {
        Frag bn;
        if (j < 7) {
            const _Float16* vr = vbase + (size_t)((j + 1) * 16 + mrow) * TT + s0 + half * 8;
            bn.u[0] = *(const uint4*)(vr);
            bn.u[1] = *(const uint4*)(vr + 16);
        }
        acc[j] = WMMA_F16(pf.h, bf.h, acc[j]);
        if (j < 7) bf = bn;
    }
    // next step's LDS writes are ordered by barrier 1 of the next iteration
}

// ---------------------------------------------------------------------------
// Causal flash attention, single head, d=256, scale=1/16.
// One 2-wave block per 16 query rows; unmasked full iterations + masked tail.
// ---------------------------------------------------------------------------
__global__ __launch_bounds__(64) void flash_attn(const _Float16* __restrict__ qh,
                                                 const _Float16* __restrict__ kh,
                                                 const _Float16* __restrict__ vt,
                                                 _Float16* __restrict__ attn_out) {
    __shared__ _Float16 pstage[16 * 32];
    __shared__ float redmax[2 * 16];
    __shared__ float redsum[2 * 16];

    const int tid  = threadIdx.x;
    const int w    = tid >> 5;              // wave id within block (0/1)
    const int lane = tid & 31;
    const int mrow = lane & 15;
    const int half = lane >> 4;
    const int qb   = blockIdx.x;            // 16-row query block id (global)
    const int b    = qb >> 8;               // qb / (T/16)
    const int t0   = (qb & 255) * 16;       // first query row within batch
    const int q0g  = qb * 16;               // first query row globally

    const _Float16* qrow  = qh + (size_t)(q0g + mrow) * DD;
    const _Float16* kbase = kh + (size_t)b * TT * DD;
    const _Float16* vbase = vt + (size_t)b * DD * TT + (size_t)(w * 128) * TT; // my 128 feats

    v8f acc[8] = {};                         // 16x128 f32 output accumulator
    float rmax[8], lsum[8];
#pragma unroll
    for (int v = 0; v < 8; ++v) { rmax[v] = -3.0e38f; lsum[v] = 0.0f; }

    const int nIter = (t0 + 47) >> 5;                      // ceil((t0+16)/32)
    const int nFull = (t0 >= 32) ? (((t0 - 32) >> 5) + 1) : 0;  // fully-causal steps

    int it = 0;
    for (; it < nFull; ++it) {
        const int s0 = it * 32;
        // prefetch next key/value block (global_prefetch_b8)
        __builtin_prefetch(kbase + (size_t)(s0 + 32 + mrow) * DD + half * 128, 0, 3);
        __builtin_prefetch(vbase + (size_t)(lane * 4) * TT + s0 + 32, 0, 3);
        attn_step<false>(s0, t0, w, mrow, half, qrow, kbase, vbase,
                         pstage, redmax, redsum, acc, rmax, lsum);
    }
    for (; it < nIter; ++it)
        attn_step<true>(it * 32, t0, w, mrow, half, qrow, kbase, vbase,
                        pstage, redmax, redsum, acc, rmax, lsum);

    // ---- finalize: divide by row sums, write my 128-feature half (f16) ----
#pragma unroll
    for (int j = 0; j < 8; ++j) {
#pragma unroll
        for (int v = 0; v < 8; ++v) {
            const int m = q0g + v + 8 * half;
            const float o = acc[j][v] / lsum[v];
            attn_out[(size_t)m * DD + w * 128 + j * 16 + mrow] = (_Float16)o;
        }
    }
}

// ---------------------------------------------------------------------------
// Host launcher
// ---------------------------------------------------------------------------
extern "C" void kernel_launch(void* const* d_in, const int* in_sizes, int n_in,
                              void* d_out, int out_size, void* d_ws, size_t ws_size,
                              hipStream_t stream) {
    const float* x      = (const float*)d_in[0];
    const float* anw    = (const float*)d_in[1];
    const float* mnw    = (const float*)d_in[2];
    const float* wq     = (const float*)d_in[3];
    const float* bq     = (const float*)d_in[4];
    const float* wk     = (const float*)d_in[5];
    const float* bk     = (const float*)d_in[6];
    const float* wv     = (const float*)d_in[7];
    const float* bv     = (const float*)d_in[8];
    const float* wo     = (const float*)d_in[9];
    const float* bo     = (const float*)d_in[10];
    const float* w1     = (const float*)d_in[11];
    const float* b1     = (const float*)d_in[12];
    const float* w2     = (const float*)d_in[13];
    const float* b2     = (const float*)d_in[14];
    float* out          = (float*)d_out;

    char* ws = (char*)d_ws;
    size_t off = 0;
    auto take = [&](size_t bytes) -> char* {
        char* p = ws + off;
        off += (bytes + 255) & ~(size_t)255;
        return p;
    };
    _Float16* xn    = (_Float16*)take((size_t)NT * DD * 2);
    _Float16* qh    = (_Float16*)take((size_t)NT * DD * 2);
    _Float16* kh    = (_Float16*)take((size_t)NT * DD * 2);
    _Float16* vt    = (_Float16*)take((size_t)NT * DD * 2);   // transposed [b][feat][t]
    _Float16* attn  = (_Float16*)take((size_t)NT * DD * 2);
    float*    x1    = (float*)   take((size_t)NT * DD * 4);
    _Float16* hn    = (_Float16*)take((size_t)NT * DD * 2);
    _Float16* hbuf  = (_Float16*)take((size_t)NT * HH * 2);
    _Float16* wq16  = (_Float16*)take((size_t)DD * DD * 2);
    _Float16* wk16  = (_Float16*)take((size_t)DD * DD * 2);
    _Float16* wv16  = (_Float16*)take((size_t)DD * DD * 2);
    _Float16* wo16  = (_Float16*)take((size_t)DD * DD * 2);
    _Float16* w116  = (_Float16*)take((size_t)HH * DD * 2);
    _Float16* w216  = (_Float16*)take((size_t)DD * HH * 2);

    // Weight conversions (f32 -> f16)
    auto cvt = [&](const float* s, _Float16* d, int n) {
        cvt_f32_to_f16<<<dim3((n + 255) / 256), dim3(256), 0, stream>>>(s, d, n);
    };
    cvt(wq, wq16, DD * DD);
    cvt(wk, wk16, DD * DD);
    cvt(wv, wv16, DD * DD);
    cvt(wo, wo16, DD * DD);
    cvt(w1, w116, HH * DD);
    cvt(w2, w216, DD * HH);

    // 1) attention RMSNorm
    rmsnorm_f16<<<dim3(NT), dim3(256), 0, stream>>>(x, anw, xn);

    // 2) Q, K, V projections (V stored transposed for the PV WMMA)
    gemm_wmma<0, DD><<<dim3(NT / 32, DD / 64), dim3(32), 0, stream>>>(xn, wq16, bq, nullptr, qh, nullptr, DD);
    gemm_wmma<0, DD><<<dim3(NT / 32, DD / 64), dim3(32), 0, stream>>>(xn, wk16, bk, nullptr, kh, nullptr, DD);
    gemm_wmma<1, DD><<<dim3(NT / 32, DD / 64), dim3(32), 0, stream>>>(xn, wv16, bv, nullptr, vt, nullptr, DD);

    // 3) causal flash attention (one 2-wave block per 16 query rows)
    flash_attn<<<dim3(NT / 16), dim3(64), 0, stream>>>(qh, kh, vt, attn);

    // 4) O-projection + residual  -> x1 (f32)
    gemm_wmma<2, DD><<<dim3(NT / 32, DD / 64), dim3(32), 0, stream>>>(attn, wo16, bo, x, nullptr, x1, DD);

    // 5) MLP RMSNorm
    rmsnorm_f16<<<dim3(NT), dim3(256), 0, stream>>>(x1, mnw, hn);

    // 6) MLP up + exact GELU -> h (f16)
    gemm_wmma<3, DD><<<dim3(NT / 32, HH / 64), dim3(32), 0, stream>>>(hn, w116, b1, nullptr, hbuf, nullptr, HH);

    // 7) MLP down + residual -> out (f32)
    gemm_wmma<2, HH><<<dim3(NT / 32, DD / 64), dim3(32), 0, stream>>>(hbuf, w216, b2, x1, nullptr, out, DD);
}
